// attentionHead_75488345194727
// MI455X (gfx1250) — matrix operand order repaired and checked by reading, not compiled, not run on hardware
//
#include <hip/hip_runtime.h>
#include <hip/hip_bf16.h>

typedef __attribute__((ext_vector_type(16))) _Float16     v16h;
typedef __attribute__((ext_vector_type(8)))  _Float16     v8h;
typedef __attribute__((ext_vector_type(4)))  _Float16     v4h;
typedef __attribute__((ext_vector_type(8)))  float        v8f;
typedef __attribute__((ext_vector_type(4)))  unsigned int v4u;
typedef __attribute__((ext_vector_type(8)))  int          v8i;
typedef __attribute__((ext_vector_type(4)))  int          v4i;

#define WMMA_F16(A, B, C) \
  __builtin_amdgcn_wmma_f32_16x16x32_f16(false, (A), false, (B), (short)0, (C), false, false)

static __device__ __forceinline__ v16h cat8(v8h lo, v8h hi) {
  return __builtin_shufflevector(lo, hi, 0,1,2,3,4,5,6,7,8,9,10,11,12,13,14,15);
}

constexpr int Bsz = 8, Lsz = 2048, Nsz = 1024, Hsz = 64;

#if __has_builtin(__builtin_amdgcn_tensor_load_to_lds) && \
    __has_builtin(__builtin_amdgcn_s_wait_tensorcnt)
#define HAVE_TDM 1
#endif

// ---------------------------------------------------------------------------
// TDM helpers: 1-D contiguous tile -> LDS.  Descriptor per CDNA5 ISA:
// group0 = {count=1, lds_addr, global_addr[56:0], type=2}
// group1 = {mask|data_size|pad cfg, tensor dims, tile_dim0, stride0}
// ---------------------------------------------------------------------------

// 32 rows x 64 halves (4KB contiguous) -> LDS rows padded to 72 halves via
// TDM pad engine (pad_interval: 32 DWORDs (code 4), pad_amount: 4 DWORDs
// (code 3)).  data_size=2B (code 1).  tensor_dim0 = 1,048,576 (hi16=16).
static __device__ __forceinline__ void stage_kv_rows(const _Float16* gsrc,
                                                     _Float16* lds, int lane) {
#ifdef HAVE_TDM
  (void)lane;
  const unsigned long long ga = (unsigned long long)(uintptr_t)gsrc;
  const unsigned ldsoff = (unsigned)(uintptr_t)lds;
  v4u g0 = { 1u, ldsoff, (unsigned)(ga & 0xFFFFFFFFu),
             (unsigned)((ga >> 32) & 0x1FFFFFFu) | (2u << 30) };
  v8i g1 = { (int)((1u << 16) | (1u << 20) | (4u << 22) | (3u << 25)),
             0,
             (int)(16u | (1u << 16)),      // tensor_dim0 hi, tensor_dim1 = 1
             (int)(2048u << 16),           // tile_dim0 = 2048 halves
             0, 2048, 0, 0 };
  v4i z4 = { 0, 0, 0, 0 };
  v8i z8 = { 0, 0, 0, 0, 0, 0, 0, 0 };
  __builtin_amdgcn_tensor_load_to_lds(g0, g1, z4, z4, z8, 0);
#else
  const _Float16* src = gsrc + lane * 64;
  _Float16* dst = lds + lane * 72;
#pragma unroll
  for (int i = 0; i < 8; ++i)
    *(v8h*)(dst + i * 8) = *(const v8h*)(src + i * 8);
#endif
}

template <int N>
static __device__ __forceinline__ void wait_stage() {
#ifdef HAVE_TDM
  __builtin_amdgcn_s_wait_tensorcnt(N);    // immediate operand required
#endif
}

// ---------------------------------------------------------------------------
// Kernel 0: transpose weights to f16 [H][N] so WMMA B columns are contiguous.
// ---------------------------------------------------------------------------
__global__ __launch_bounds__(256) void wtrans_kernel(
    const float* __restrict__ Wk, const float* __restrict__ Wq,
    const float* __restrict__ Wv, _Float16* __restrict__ Wt)
{
  const int idx = blockIdx.x * 256 + threadIdx.x;   // h*1024 + n
  if (idx >= Hsz * Nsz) return;
  const int h = idx >> 10;
  const int n = idx & 1023;
  Wt[idx]                 = (_Float16)Wq[n * Hsz + h];
  Wt[Hsz * Nsz + idx]     = (_Float16)Wk[n * Hsz + h];
  Wt[2 * Hsz * Nsz + idx] = (_Float16)Wv[n * Hsz + h];
}

// ---------------------------------------------------------------------------
// Kernel 1: QKV projection.  x slab (64KB f32) DMA'd to LDS by the TDM, then
// one cooperative f32->f16 pass into a second LDS buffer so the per-chunk A
// operand is two pure ds_load_b128 (no redundant per-wave conversions).
// ---------------------------------------------------------------------------
__global__ __launch_bounds__(128) void qkv_proj_kernel(
    const float* __restrict__ x,
    const _Float16* __restrict__ Wtq, const _Float16* __restrict__ Wtk,
    const _Float16* __restrict__ Wtv,
    _Float16* __restrict__ Qw, _Float16* __restrict__ Kw, _Float16* __restrict__ Vw)
{
  __shared__ __align__(16) float    xs[16 * Nsz];   // 64 KB f32 slab
  __shared__ __align__(16) _Float16 xh[16 * Nsz];   // 32 KB f16 slab

  const size_t rowbase = (size_t)blockIdx.x * 16;
  const float* xin = x + rowbase * Nsz;

#ifdef HAVE_TDM
  if (threadIdx.x < 32) {                           // wave 0 issues the DMA
    const unsigned long long gaddr = (unsigned long long)(uintptr_t)xin;
    const unsigned ldsoff = (unsigned)(uintptr_t)&xs[0];
    v4u g0 = { 1u, ldsoff, (unsigned)(gaddr & 0xFFFFFFFFu),
               (unsigned)((gaddr >> 32) & 0x1FFFFFFu) | (2u << 30) };
    // data_size=4B (code 2); tensor_dim0 = B*L*N = 16,777,216 (hi16=256);
    // tile_dim0 = 16384 f32 = 64KB contiguous; no padding.
    v8i g1 = { (int)(2u << 16),
               0,
               (int)(256u | (1u << 16)),
               (int)(16384u << 16),
               0, 16384, 0, 0 };
    v4i z4 = { 0, 0, 0, 0 };
    v8i z8 = { 0, 0, 0, 0, 0, 0, 0, 0 };
    __builtin_amdgcn_tensor_load_to_lds(g0, g1, z4, z4, z8, 0);
    __builtin_amdgcn_s_wait_tensorcnt(0);
  }
  __syncthreads();
#else
  for (int i = threadIdx.x; i < 16 * Nsz / 4; i += 128)
    ((float4*)xs)[i] = ((const float4*)xin)[i];
  __syncthreads();
#endif

  // One-pass f32 -> f16 conversion (shared by all 4 waves).
  for (int i = threadIdx.x; i < 16 * Nsz / 4; i += 128) {
    float4 f = ((const float4*)xs)[i];
    v4h h = { (_Float16)f.x, (_Float16)f.y, (_Float16)f.z, (_Float16)f.w };
    *((v4h*)xh + i) = h;
  }
  __syncthreads();

  const int wave = threadIdx.x >> 5;                // 0..3 -> H tile
  const int lane = threadIdx.x & 31;
  const int cidx = lane & 15;
  const int hi   = lane >> 4;
  const int colg = wave * 16 + cidx;

  v8f aq = {}, ak = {}, av = {};

  const _Float16* wqc = Wtq + (size_t)colg * Nsz;   // B column = Wt row
  const _Float16* wkc = Wtk + (size_t)colg * Nsz;
  const _Float16* wvc = Wtv + (size_t)colg * Nsz;

  for (int kc = 0; kc < Nsz / 32; ++kc) {
    const int k0 = kc * 32;
    // A operand (contiguous layout): row=cidx, K=0..15/16..31
    const _Float16* ar = xh + cidx * Nsz + k0;
    v16h A = cat8(*(const v8h*)(ar + 16 * hi), *(const v8h*)(ar + 16 * hi + 8));
    // B operands (contiguous layout): lanes 0-15 K=0..15, lanes 16-31 K=16..31
    const int kb = k0 + 16 * hi;
    v16h Bq = cat8(*(const v8h*)(wqc + kb), *(const v8h*)(wqc + kb + 8));
    v16h Bk = cat8(*(const v8h*)(wkc + kb), *(const v8h*)(wkc + kb + 8));
    v16h Bv = cat8(*(const v8h*)(wvc + kb), *(const v8h*)(wvc + kb + 8));

    aq = WMMA_F16(A, Bq, aq);
    ak = WMMA_F16(A, Bk, ak);
    av = WMMA_F16(A, Bv, av);
  }

#pragma unroll
  for (int v = 0; v < 8; ++v) {
    const size_t o = (rowbase + v + 8 * hi) * Hsz + colg;
    Qw[o] = (_Float16)aq[v];
    Kw[o] = (_Float16)ak[v];
    Vw[o] = (_Float16)av[v];
  }
}

// ---------------------------------------------------------------------------
// Kernel 2: causal flash attention, TDM double-buffered.  One wave per 16-row
// q tile.  Per 32-wide kv chunk the wave TDM-loads next chunk's K and V (4KB
// contiguous each) into the alternate per-wave LDS buffer (pad engine makes
// 72-half rows: aligned + bank-spread), waits tensorcnt<=2 for the current
// chunk, then runs 4 score WMMAs + online softmax + 4 PV WMMAs from LDS.
// ---------------------------------------------------------------------------
__global__ __launch_bounds__(128) void flash_attn_kernel(
    const _Float16* __restrict__ Qw, const _Float16* __restrict__ Kw,
    const _Float16* __restrict__ Vw, float* __restrict__ out)
{
  __shared__ __align__(16) _Float16 pbuf[4][16][32];
  __shared__ __align__(16) _Float16 kvbuf[4][2][2][32 * 72];  // [wave][buf][K/V]

  const int wave = threadIdx.x >> 5;
  const int lane = threadIdx.x & 31;
  const int cidx = lane & 15;
  const int hi   = lane >> 4;

  const int qt = blockIdx.x * 4 + wave;        // global q tile 0..1023
  const int b  = qt >> 7;                      // 128 tiles per batch
  const int q0 = (qt & 127) * 16;
  const size_t bL = (size_t)b * Lsz;

  // Q as two 16x32 A operands (K-dim = H = 64)
  const _Float16* qrow = Qw + (bL + q0 + cidx) * Hsz;
  v16h qa0 = cat8(*(const v8h*)(qrow + 16 * hi),      *(const v8h*)(qrow + 16 * hi + 8));
  v16h qa1 = cat8(*(const v8h*)(qrow + 32 + 16 * hi), *(const v8h*)(qrow + 40 + 16 * hi));

  v8f O0 = {}, O1 = {}, O2 = {}, O3 = {};
  float mrow[8], lrow[8];
#pragma unroll
  for (int v = 0; v < 8; ++v) { mrow[v] = -__builtin_inff(); lrow[v] = 0.0f; }

  const float scale = 0.03125f;                // N^-0.5 = 1/32
  const int qend    = q0 + 16;
  const int nchunks = (qend + 31) / 32;

  // Prologue: stage chunk 0 into buffer 0.
  stage_kv_rows(Kw + bL * Hsz, &kvbuf[wave][0][0][0], lane);
  stage_kv_rows(Vw + bL * Hsz, &kvbuf[wave][0][1][0], lane);

  int cur = 0;
  for (int c = 0; c < nchunks; ++c) {
    const int kv0 = c * 32;
    if (c + 1 < nchunks) {                     // async prefetch of next chunk
      stage_kv_rows(Kw + (bL + kv0 + 32) * Hsz, &kvbuf[wave][cur ^ 1][0][0], lane);
      stage_kv_rows(Vw + (bL + kv0 + 32) * Hsz, &kvbuf[wave][cur ^ 1][1][0], lane);
      wait_stage<2>();                         // in-order: chunk c complete
    } else {
      wait_stage<0>();
    }
    const _Float16* kb = &kvbuf[wave][cur][0][0];
    const _Float16* vb = &kvbuf[wave][cur][1][0];

    // ---- scores: two 16x16 tiles, K-dim 64 each, B from LDS ----
    v8f acc0 = {}, acc1 = {};
    {
      const _Float16* kr = kb + cidx * 72;     // B col = K row
      v16h b0 = cat8(*(const v8h*)(kr + 16 * hi),      *(const v8h*)(kr + 16 * hi + 8));
      v16h b1 = cat8(*(const v8h*)(kr + 32 + 16 * hi), *(const v8h*)(kr + 40 + 16 * hi));
      acc0 = WMMA_F16(qa0, b0, acc0);
      acc0 = WMMA_F16(qa1, b1, acc0);
    }
    {
      const _Float16* kr = kb + (16 + cidx) * 72;
      v16h b0 = cat8(*(const v8h*)(kr + 16 * hi),      *(const v8h*)(kr + 16 * hi + 8));
      v16h b1 = cat8(*(const v8h*)(kr + 32 + 16 * hi), *(const v8h*)(kr + 40 + 16 * hi));
      acc1 = WMMA_F16(qa0, b0, acc1);
      acc1 = WMMA_F16(qa1, b1, acc1);
    }

    // ---- mask + online softmax ----
#pragma unroll
    for (int v = 0; v < 8; ++v) {
      const int qrg = q0 + v + 8 * hi;
      float s0 = acc0[v] * scale;
      float s1 = acc1[v] * scale;
      if (kv0 + cidx > qrg)      s0 = -__builtin_inff();
      if (kv0 + 16 + cidx > qrg) s1 = -__builtin_inff();

      float cm = fmaxf(s0, s1);
      cm = fmaxf(cm, __shfl_xor(cm, 1, 32));
      cm = fmaxf(cm, __shfl_xor(cm, 2, 32));
      cm = fmaxf(cm, __shfl_xor(cm, 4, 32));
      cm = fmaxf(cm, __shfl_xor(cm, 8, 32));
      const float nm    = fmaxf(mrow[v], cm);
      const float alpha = __expf(mrow[v] - nm);
      mrow[v] = nm;

      const float p0 = __expf(s0 - nm);
      const float p1 = __expf(s1 - nm);
      float rs = p0 + p1;
      rs += __shfl_xor(rs, 1, 32);
      rs += __shfl_xor(rs, 2, 32);
      rs += __shfl_xor(rs, 4, 32);
      rs += __shfl_xor(rs, 8, 32);
      lrow[v] = lrow[v] * alpha + rs;

      O0[v] *= alpha; O1[v] *= alpha; O2[v] *= alpha; O3[v] *= alpha;

      const int r = v + 8 * hi;
      pbuf[wave][r][cidx]      = (_Float16)p0;   // D-layout -> row-major LDS
      pbuf[wave][r][cidx + 16] = (_Float16)p1;
    }

    // ---- P as 16x32 A operand (contiguous A layout) ----
    const _Float16* pr = &pbuf[wave][cidx][0];
    v16h pa = cat8(*(const v8h*)(pr + 16 * hi), *(const v8h*)(pr + 16 * hi + 8));

    // ---- V as four 32x16 B operands, columns from padded LDS rows ----
    v16h bv0, bv1, bv2, bv3;
#pragma unroll
    for (int j = 0; j < 16; ++j) {
      const _Float16* vp = vb + (j + 16 * hi) * 72 + cidx;
      bv0[j] = vp[0];  bv1[j] = vp[16];
      bv2[j] = vp[32]; bv3[j] = vp[48];
    }
    O0 = WMMA_F16(pa, bv0, O0);
    O1 = WMMA_F16(pa, bv1, O1);
    O2 = WMMA_F16(pa, bv2, O2);
    O3 = WMMA_F16(pa, bv3, O3);

    cur ^= 1;
  }

  // ---- epilogue: divide by row sum, store fp32 ----
#pragma unroll
  for (int v = 0; v < 8; ++v) {
    const float inv = 1.0f / lrow[v];
    const size_t o = (bL + q0 + v + 8 * hi) * Hsz + cidx;
    out[o]      = O0[v] * inv;
    out[o + 16] = O1[v] * inv;
    out[o + 32] = O2[v] * inv;
    out[o + 48] = O3[v] * inv;
  }
}

// ---------------------------------------------------------------------------
extern "C" void kernel_launch(void* const* d_in, const int* in_sizes, int n_in,
                              void* d_out, int out_size, void* d_ws, size_t ws_size,
                              hipStream_t stream) {
  (void)in_sizes; (void)n_in; (void)out_size; (void)ws_size;
  const float* x  = (const float*)d_in[0];
  const float* Wk = (const float*)d_in[1];
  const float* Wq = (const float*)d_in[2];
  const float* Wv = (const float*)d_in[3];
  float* out = (float*)d_out;

  const size_t elems = (size_t)Bsz * Lsz * Hsz;     // 1M halves per tensor
  _Float16* Qw  = (_Float16*)d_ws;
  _Float16* Kw  = Qw + elems;
  _Float16* Vw  = Kw + elems;
  _Float16* Wt  = Vw + elems;                       // 3 x 64 x 1024 halves
  _Float16* Wtq = Wt;
  _Float16* Wtk = Wt + (size_t)Hsz * Nsz;
  _Float16* Wtv = Wt + 2 * (size_t)Hsz * Nsz;

  wtrans_kernel<<<(Hsz * Nsz) / 256, 256, 0, stream>>>(Wk, Wq, Wv, Wt);
  qkv_proj_kernel<<<(Bsz * Lsz) / 16, 128, 0, stream>>>(x, Wtq, Wtk, Wtv, Qw, Kw, Vw);
  flash_attn_kernel<<<(Bsz * Lsz) / (16 * 4), 128, 0, stream>>>(Qw, Kw, Vw, out);
}
